// GraphTransformerModel_41240275976660
// MI455X (gfx1250) — matrix operand order repaired and checked by reading, not compiled, not run on hardware
//
#include <hip/hip_runtime.h>

typedef _Float16 v16h __attribute__((ext_vector_type(16)));
typedef _Float16 v8h  __attribute__((ext_vector_type(8)));
typedef float    v8f  __attribute__((ext_vector_type(8)));

#define NN   256
#define DD   128
#define EDIM 16
#define HH   8
#define HD   16

// ---------------------------------------------------------------------------
// Pack helpers: one-shot f32 -> f16 relayouts so WMMA operand loads in the hot
// loops are contiguous vector loads with zero in-loop conversion.
// ---------------------------------------------------------------------------
__global__ __launch_bounds__(256) void pack_f16(const float* __restrict__ src,
                                                _Float16* __restrict__ dst, int n) {
  const int i = blockIdx.x * 256 + threadIdx.x;
  if (i < n) dst[i] = (_Float16)src[i];
}

// W (128x128 row-major f32) -> WT (f16, WT[n*128+k] = W[k*128+n])
__global__ __launch_bounds__(256) void pack_wt(const float* __restrict__ W,
                                               _Float16* __restrict__ WT) {
  const int i = blockIdx.x * 256 + threadIdx.x;  // 0..16383
  const int n = i >> 7, k = i & 127;
  WT[n * DD + k] = (_Float16)W[k * DD + n];
}

// k,v (B,N,D f32) -> KH (bh,j,16) f16 contiguous and VT (bh,d,j) f16 transposed
__global__ __launch_bounds__(256) void pack_kv(const float* __restrict__ K,
                                               const float* __restrict__ V,
                                               _Float16* __restrict__ KH,
                                               _Float16* __restrict__ VT) {
  const int idx = blockIdx.x * 256 + threadIdx.x;  // bh*256 + j
  const int bh = idx >> 8, j = idx & 255;
  const int b = bh >> 3, h = bh & 7;
  const float* kr = K + (b * NN + j) * DD + h * HD;
  const float* vr = V + (b * NN + j) * DD + h * HD;
  _Float16* ko = KH + idx * 16;
#pragma unroll
  for (int d = 0; d < 16; ++d) {
    ko[d] = (_Float16)kr[d];
    VT[(bh * 16 + d) * NN + j] = (_Float16)vr[d];
  }
}

// ---------------------------------------------------------------------------
// Kernel A: O(512x128) = X @ W + bias via WMMA f16->f32; operands pre-packed.
// One wave per 16x16 output tile; K=128 in 4 chunks of 32.
// ---------------------------------------------------------------------------
__global__ __launch_bounds__(32) void proj_gemm(const _Float16* __restrict__ XH,
                                                const _Float16* __restrict__ WT,
                                                const float* __restrict__ bias,
                                                float* __restrict__ O) {
  const int rt   = blockIdx.x;     // row tile 0..31
  const int ct   = blockIdx.y;     // col tile 0..7
  const int lane = threadIdx.x;
  const int mrow = lane & 15;
  const int sel  = lane >> 4;      // lane half
  v8f acc = {0.f, 0.f, 0.f, 0.f, 0.f, 0.f, 0.f, 0.f};
  const int col = ct * 16 + mrow;
#pragma unroll
  for (int k0 = 0; k0 < DD; k0 += 32) {
    // A tile 16x32: lanes 0-15 hold K {k0..k0+15}; lanes 16-31 hold
    // K {k0+16..31}; row = mrow.  Two contiguous 16B loads.
    const _Float16* xr = XH + (rt * 16 + mrow) * DD + k0 + sel * 16;
    const v8h alo = *(const v8h*)(xr);
    const v8h ahi = *(const v8h*)(xr + 8);
    v16h a;
#pragma unroll
    for (int t = 0; t < 8; ++t) { a[t] = alo[t]; a[8 + t] = ahi[t]; }
    // B tile 32x16: per-lane 16 consecutive K for N=col. One 32B load.
    const v16h bm = *(const v16h*)(WT + col * DD + k0 + sel * 16);
    acc = __builtin_amdgcn_wmma_f32_16x16x32_f16(false, a, false, bm,
                                                 (short)0, acc, false, false);
  }
  const float bv    = bias[col];
  const int   rbase = rt * 16 + sel * 8;
#pragma unroll
  for (int r = 0; r < 8; ++r)
    O[(rbase + r) * DD + col] = acc[r] + bv;
}

// ---------------------------------------------------------------------------
// Kernel W: wq[bh,n,c] = sum_d We[c, h*16+d] * q[b,n,h*16+d]   (tiny)
// ---------------------------------------------------------------------------
__global__ __launch_bounds__(256) void wq_kernel(const float* __restrict__ Q,
                                                 const float* __restrict__ We,
                                                 float* __restrict__ WQE) {
  const int idx = blockIdx.x * 256 + threadIdx.x;  // bh*256 + n
  const int bh = idx >> 8, n = idx & 255;
  const int b = bh >> 3, h = bh & 7;
  const float* qr = Q + (b * NN + n) * DD + h * HD;
  float qv[16];
#pragma unroll
  for (int d = 0; d < 16; ++d) qv[d] = qr[d];
  float* o = WQE + idx * 16;
#pragma unroll 4
  for (int c = 0; c < 16; ++c) {
    const float* wr = We + c * DD + h * HD;
    float s = 0.f;
#pragma unroll
    for (int d = 0; d < 16; ++d) s += wr[d] * qv[d];
    o[c] = s;
  }
}

// ---------------------------------------------------------------------------
// Kernel S: fused attention. One workgroup = (b, itile); one wave = one head.
// sim = q.kT (WMMA, K padded 16->32) + edge-dot (VALU); exp*adj; accumulate
// rowsum l, weighted-edge ea, and attn@v (WMMA). Epilogue adds ea@We_h + be.
// ---------------------------------------------------------------------------
__global__ __launch_bounds__(256) void attn_kernel(
    const float* __restrict__ Q, const _Float16* __restrict__ KH,
    const _Float16* __restrict__ VT, const float* __restrict__ WQE,
    const float* __restrict__ E, const int* __restrict__ ADJ,
    const float* __restrict__ We, const float* __restrict__ be,
    float* __restrict__ T) {
  __shared__ float SH[8 * 1024];  // per wave: 256 wq | 512 p | 256 ea
  const int b     = blockIdx.x >> 4;
  const int itile = blockIdx.x & 15;
  const int wid   = threadIdx.x >> 5;  // head
  const int lane  = threadIdx.x & 31;
  const int h  = wid;
  const int bh = b * HH + h;
  const int i0 = itile * 16;

  float* swq = SH + wid * 1024;
  float* sp  = swq + 256;
  float* sea = sp + 512;

  // stage this wave's wq tile (16 rows x 16 coeffs)
#pragma unroll
  for (int t = 0; t < 8; ++t) {
    const int idx = t * 32 + lane;
    swq[idx] = WQE[(bh * NN + i0 + (idx >> 4)) * 16 + (idx & 15)];
  }

  const int mrow = lane & 15;
  const int sel  = lane >> 4;

  v16h zv;
#pragma unroll
  for (int t = 0; t < 16; ++t) zv[t] = (_Float16)0.f;

  // A_q operand: 16x32, K 0..15 = q head dims, K 16..31 = 0 (pad)
  v16h aq;
  {
    const float* qr = Q + (b * NN + i0 + mrow) * DD + h * HD;
#pragma unroll
    for (int t = 0; t < 8; ++t) aq[t]     = (sel == 0) ? (_Float16)qr[t] : (_Float16)0.f;
#pragma unroll
    for (int t = 0; t < 8; ++t) aq[8 + t] = (sel == 0) ? (_Float16)qr[8 + t] : (_Float16)0.f;
  }

  v8f   acc     = {0.f, 0.f, 0.f, 0.f, 0.f, 0.f, 0.f, 0.f};
  float lsum[8] = {0.f, 0.f, 0.f, 0.f, 0.f, 0.f, 0.f, 0.f};
  float ea[8]   = {0.f, 0.f, 0.f, 0.f, 0.f, 0.f, 0.f, 0.f};

  for (int jc = 0; jc < 8; ++jc) {
    const int jbase = jc * 32;
#pragma unroll
    for (int st = 0; st < 2; ++st) {
      const int jb = jbase + st * 16;
      // B_k: lanes 0-15 hold K=0..15 (k head dims) for N=j; lanes 16-31 hold
      // the padded K=16..31 -> zeros.  One contiguous 32B load + select.
      const v16h kv = *(const v16h*)(KH + (bh * NN + jb + mrow) * 16);
      const v16h bk = (sel == 0) ? kv : zv;
      v8f zc = {0.f, 0.f, 0.f, 0.f, 0.f, 0.f, 0.f, 0.f};
      v8f s = __builtin_amdgcn_wmma_f32_16x16x32_f16(false, aq, false, bk,
                                                     (short)0, zc, false, false);
      // edge-dot + scale + exp*adj; p tile -> LDS (C layout: i=r+8*sel, j=mrow)
      const int jj = jb + mrow;
#pragma unroll
      for (int r = 0; r < 8; ++r) {
        const int i = r + sel * 8;
        const float* er = E + ((b * NN + i0 + i) * NN + jj) * EDIM;
        if (jc < 7) __builtin_prefetch(er + 32 * EDIM, 0, 0);
        float ed = 0.f;
#pragma unroll
        for (int c = 0; c < 16; ++c) ed += swq[i * 16 + c] * er[c];
        const float sv = (s[r] + ed) * 0.25f;  // 1/sqrt(HD)
        const float av = (float)ADJ[(b * NN + i0 + i) * NN + jj];
        const float p  = __expf(sv) * av;      // shift-invariant softmax, m=0
        lsum[r] += p;
        sp[i * 32 + st * 16 + mrow] = p;
      }
    }
    // ea[i,c] += sum_j p[i,j] * E[i,j,c]   (lane: i=mrow, c-half by sel)
    {
      const int coff = sel * 8;
#pragma unroll 4
      for (int j = 0; j < 32; ++j) {
        const float pv = sp[mrow * 32 + j];
        const float* er =
            E + ((b * NN + i0 + mrow) * NN + jbase + j) * EDIM + coff;
#pragma unroll
        for (int c = 0; c < 8; ++c) ea[c] += pv * er[c];
      }
    }
    // A_p operand (16x32 over this j-chunk) from LDS
    v16h ap;
#pragma unroll
    for (int t = 0; t < 8; ++t) ap[t]     = (_Float16)sp[mrow * 32 + sel * 16 + t];
#pragma unroll
    for (int t = 0; t < 8; ++t) ap[8 + t] = (_Float16)sp[mrow * 32 + sel * 16 + 8 + t];
    // B_v operand: per-lane 16 consecutive K(=j) for N=d. One 32B load from VT.
    const v16h bv =
        *(const v16h*)(VT + (bh * 16 + mrow) * NN + jbase + sel * 16);
    acc = __builtin_amdgcn_wmma_f32_16x16x32_f16(false, ap, false, bv,
                                                 (short)0, acc, false, false);
  }

  // reduce row sums across the 16 lanes of each half
#pragma unroll
  for (int r = 0; r < 8; ++r) {
    float v = lsum[r];
#pragma unroll
    for (int m = 1; m < 16; m <<= 1) v += __shfl_xor(v, m, 16);
    lsum[r] = v;
  }
  // share ea through LDS (writer mapping != reader mapping, same wave)
  {
    const int coff = sel * 8;
#pragma unroll
    for (int c = 0; c < 8; ++c) sea[mrow * 16 + coff + c] = ea[c];
  }
  // epilogue: out = (acc + ea@We_h)/l + be_h * (l>0)
  {
    const int d = mrow;
    float wcol[16];
#pragma unroll
    for (int c = 0; c < 16; ++c) wcol[c] = We[c * DD + h * HD + d];
    const float bev = be[h * HD + d];
#pragma unroll
    for (int r = 0; r < 8; ++r) {
      const int i = r + sel * 8;
      float extra = 0.f;
#pragma unroll
      for (int c = 0; c < 16; ++c) extra += sea[i * 16 + c] * wcol[c];
      const float lraw = lsum[r];
      const float l    = (lraw == 0.f) ? 1.f : lraw;
      const float satt = (lraw == 0.f) ? 0.f : 1.f;
      T[(b * NN + i0 + i) * DD + h * HD + d] = (acc[r] + extra) / l + bev * satt;
    }
  }
}

// ---------------------------------------------------------------------------
// Kernel G: gated residual + LayerNorm + ReLU. One row per 128-thread block.
// ---------------------------------------------------------------------------
__global__ __launch_bounds__(128) void gate_ln(const float* __restrict__ T,
                                               const float* __restrict__ R,
                                               const float* __restrict__ Wp,
                                               const float* __restrict__ gamma,
                                               const float* __restrict__ beta,
                                               float* __restrict__ Xout) {
  __shared__ float red[128];
  const int row = blockIdx.x;
  const int d   = threadIdx.x;
  const float t = T[row * DD + d];
  const float r = R[row * DD + d];
  red[d] = t * Wp[d] + r * Wp[DD + d] + (t - r) * Wp[2 * DD + d];
  __syncthreads();
  for (int s = 64; s > 0; s >>= 1) {
    if (d < s) red[d] += red[d + s];
    __syncthreads();
  }
  const float g = 1.f / (1.f + __expf(-red[0]));
  __syncthreads();
  const float mixed = t * g + r * (1.f - g);
  red[d] = mixed;
  __syncthreads();
  for (int s = 64; s > 0; s >>= 1) {
    if (d < s) red[d] += red[d + s];
    __syncthreads();
  }
  const float mean = red[0] * (1.f / 128.f);
  __syncthreads();
  const float dv = mixed - mean;
  red[d] = dv * dv;
  __syncthreads();
  for (int s = 64; s > 0; s >>= 1) {
    if (d < s) red[d] += red[d + s];
    __syncthreads();
  }
  const float var = red[0] * (1.f / 128.f);
  const float y = dv * __frsqrt_rn(var + 1e-5f) * gamma[d] + beta[d];
  Xout[row * DD + d] = fmaxf(y, 0.f);
}

// ---------------------------------------------------------------------------
// Host driver. Param index map (setup_inputs dict order): 0 nodes, 1 edges,
// 2 adjacency, then per block i (base = 3+13*i):
// Wq,Wk,Wv,We,Wr,Wp,bq,bk,bv,be,br,gamma,beta
// ---------------------------------------------------------------------------
extern "C" void kernel_launch(void* const* d_in, const int* in_sizes, int n_in,
                              void* d_out, int out_size, void* d_ws, size_t ws_size,
                              hipStream_t stream) {
  const float* nodes = (const float*)d_in[0];
  const float* edges = (const float*)d_in[1];
  const int*   adj   = (const int*)d_in[2];

  float* ws  = (float*)d_ws;
  float* qb  = ws;            // 512*128 f32
  float* kb  = qb  + 65536;
  float* vb  = kb  + 65536;
  float* rb  = vb  + 65536;   // res2
  float* wqb = rb  + 65536;   // 16*256*16 f32
  float* tb  = wqb + 65536;   // attention out
  float* xb  = tb  + 65536;   // x between blocks
  _Float16* xh = (_Float16*)(xb + 65536);  // 512*128 f16
  _Float16* kh = xh + 65536;               // (bh, j, 16) f16
  _Float16* vt = kh + 65536;               // (bh, d, j) f16
  _Float16* wt0 = vt + 65536;              // 4 x 128*128 f16 (Wq,Wk,Wv,Wr)^T
  _Float16* wt1 = wt0 + 16384;
  _Float16* wt2 = wt1 + 16384;
  _Float16* wt3 = wt2 + 16384;

  for (int i = 0; i < 2; ++i) {
    const int base = 3 + 13 * i;
    const float* Wq = (const float*)d_in[base + 0];
    const float* Wk = (const float*)d_in[base + 1];
    const float* Wv = (const float*)d_in[base + 2];
    const float* We = (const float*)d_in[base + 3];
    const float* Wr = (const float*)d_in[base + 4];
    const float* Wp = (const float*)d_in[base + 5];
    const float* bq = (const float*)d_in[base + 6];
    const float* bk = (const float*)d_in[base + 7];
    const float* bv = (const float*)d_in[base + 8];
    const float* be = (const float*)d_in[base + 9];
    const float* br = (const float*)d_in[base + 10];
    const float* gamma = (const float*)d_in[base + 11];
    const float* beta  = (const float*)d_in[base + 12];

    const float* x = (i == 0) ? nodes : xb;
    // operand packing (one pass each)
    hipLaunchKernelGGL(pack_f16, dim3(256), dim3(256), 0, stream, x, xh, 65536);
    hipLaunchKernelGGL(pack_wt, dim3(64), dim3(256), 0, stream, Wq, wt0);
    hipLaunchKernelGGL(pack_wt, dim3(64), dim3(256), 0, stream, Wk, wt1);
    hipLaunchKernelGGL(pack_wt, dim3(64), dim3(256), 0, stream, Wv, wt2);
    hipLaunchKernelGGL(pack_wt, dim3(64), dim3(256), 0, stream, Wr, wt3);
    // projections
    const dim3 gA(32, 8);
    hipLaunchKernelGGL(proj_gemm, gA, dim3(32), 0, stream, xh, wt0, bq, qb);
    hipLaunchKernelGGL(proj_gemm, gA, dim3(32), 0, stream, xh, wt1, bk, kb);
    hipLaunchKernelGGL(proj_gemm, gA, dim3(32), 0, stream, xh, wt2, bv, vb);
    hipLaunchKernelGGL(proj_gemm, gA, dim3(32), 0, stream, xh, wt3, br, rb);
    // attention precompute + fused attention
    hipLaunchKernelGGL(wq_kernel, dim3(16), dim3(256), 0, stream, qb, We, wqb);
    hipLaunchKernelGGL(pack_kv, dim3(16), dim3(256), 0, stream, kb, vb, kh, vt);
    hipLaunchKernelGGL(attn_kernel, dim3(32), dim3(256), 0, stream,
                       qb, kh, vt, wqb, edges, adj, We, be, tb);
    float* xout = (i == 1) ? (float*)d_out : xb;
    hipLaunchKernelGGL(gate_ln, dim3(512), dim3(128), 0, stream,
                       tb, rb, Wp, gamma, beta, xout);
  }
  (void)in_sizes; (void)n_in; (void)out_size; (void)ws_size;
}
